// AdvancedAttention_23922967839333
// MI455X (gfx1250) — matrix operand order repaired and checked by reading, not compiled, not run on hardware
//
#include <hip/hip_runtime.h>

typedef __bf16 bf16;
typedef __attribute__((ext_vector_type(16))) __bf16       v16bf;
typedef __attribute__((ext_vector_type(8)))  float        v8f;
typedef __attribute__((ext_vector_type(4)))  unsigned int v4u;
typedef __attribute__((ext_vector_type(4)))  unsigned int u32x4;
typedef __attribute__((ext_vector_type(8)))  int          i32x8;
typedef __attribute__((ext_vector_type(4)))  int          i32x4;

union Frag16 { v4u u[2]; v16bf v; };
union Pack8  { v4u u;    unsigned short s[8]; };

#define NB   2
#define NS   2048
#define NDM  2048
#define NH   16
#define NHD  128
#define NM   (NB*NS)

#if __has_builtin(__builtin_amdgcn_tensor_load_to_lds) && __has_builtin(__builtin_amdgcn_s_wait_tensorcnt)
#define HAVE_TDM 1
#else
#define HAVE_TDM 0
#endif

static __device__ __forceinline__ unsigned short f2bu(float x) {
  bf16 h = (bf16)x; unsigned short u; __builtin_memcpy(&u, &h, 2); return u;
}
static __device__ __forceinline__ float wsum16(float x) {
  x += __shfl_xor(x, 1, 32); x += __shfl_xor(x, 2, 32);
  x += __shfl_xor(x, 4, 32); x += __shfl_xor(x, 8, 32);
  return x;
}
static __device__ __forceinline__ float wmax16(float x) {
  x = fmaxf(x, __shfl_xor(x, 1, 32)); x = fmaxf(x, __shfl_xor(x, 2, 32));
  x = fmaxf(x, __shfl_xor(x, 4, 32)); x = fmaxf(x, __shfl_xor(x, 8, 32));
  return x;
}

#if HAVE_TDM
// Issue one 2-D TENSOR_LOAD_TO_LDS. Tensor dims == tile dims (no OOB),
// dim0 = contiguous axis, dim1 strided by stride0 elements (bf16).
// pad_interval/pad_amount: LDS row padding codes per CDNA5 D# spec.
static __device__ __forceinline__ void tdm_load_2d(
    unsigned lds_addr, const void* gaddr,
    unsigned tile0, unsigned tile1, unsigned stride0,
    unsigned pad_interval, unsigned pad_amount)
{
  unsigned long long ga = (unsigned long long)(size_t)gaddr;
  u32x4 g0;
  g0.x = 1u;                                          // count=1, user load
  g0.y = lds_addr;                                    // LDS byte address
  g0.z = (unsigned)ga;                                // global_addr[31:0]
  g0.w = ((unsigned)(ga >> 32) & 0x01FFFFFFu) | 0x80000000u; // addr[56:32]|type=2
  i32x8 g1;
  g1[0] = (int)((1u << 16) |                          // data_size = 2 bytes
                (1u << 20) |                          // pad_enable
                (pad_interval << 22) | (pad_amount << 25));
  g1[1] = (int)((tile0 & 0xFFFFu) << 16);             // tensor_dim0[15:0]
  g1[2] = (int)((tile0 >> 16) | ((tile1 & 0xFFFFu) << 16)); // dim0 hi | dim1 lo
  g1[3] = (int)((tile1 >> 16) | (tile0 << 16));       // dim1 hi | tile_dim0
  g1[4] = (int)tile1;                                 // tile_dim1 (tile_dim2=0)
  g1[5] = (int)stride0;                               // tensor_dim0_stride lo
  g1[6] = 0; g1[7] = 0;
  i32x4 gz = {0, 0, 0, 0};
#if defined(__clang_major__) && (__clang_major__ >= 23)
  i32x8 gz8 = {0, 0, 0, 0, 0, 0, 0, 0};
  __builtin_amdgcn_tensor_load_to_lds(g0, g1, gz, gz, gz8, 0);
#else
  __builtin_amdgcn_tensor_load_to_lds(g0, g1, gz, gz, 0);
#endif
}
#endif

// ---------------------------------------------------------------------------
// Setup 0a: hidden_states f32 -> bf16.
// ---------------------------------------------------------------------------
__global__ __launch_bounds__(256)
void cvt_hs(const float* __restrict__ hs, unsigned short* __restrict__ hsb) {
  size_t i = ((size_t)blockIdx.x * 256 + threadIdx.x) * 8;
  const float4* p = (const float4*)(hs + i);
  float4 a = p[0], b = p[1];
  Pack8 w;
  w.s[0]=f2bu(a.x); w.s[1]=f2bu(a.y); w.s[2]=f2bu(a.z); w.s[3]=f2bu(a.w);
  w.s[4]=f2bu(b.x); w.s[5]=f2bu(b.y); w.s[6]=f2bu(b.z); w.s[7]=f2bu(b.w);
  *(v4u*)(hsb + i) = w.u;
}

// ---------------------------------------------------------------------------
// Setup 0b: W (n,k) f32 -> Wt (k,n) bf16 (B-major for WMMA).
// ---------------------------------------------------------------------------
__global__ __launch_bounds__(256)
void cvt_w(const float* __restrict__ Wq, const float* __restrict__ Wk,
           const float* __restrict__ Wv, unsigned short* __restrict__ Wt) {
  const int z = blockIdx.z;
  const float* __restrict__ W = (z == 0) ? Wq : (z == 1) ? Wk : Wv;
  unsigned short* __restrict__ o = Wt + (size_t)z * NDM * NDM;
  __shared__ __align__(16) bf16 tile[64][72];
  const int n0 = blockIdx.x * 64, k0 = blockIdx.y * 64;
  const int t = threadIdx.x, r = t >> 2, c = (t & 3) * 16;
  {
    const float4* p = (const float4*)(W + (size_t)(n0 + r) * NDM + k0 + c);
    #pragma unroll
    for (int v = 0; v < 4; ++v) {
      float4 f = p[v]; int cc = c + v * 4;
      tile[r][cc+0] = (bf16)f.x; tile[r][cc+1] = (bf16)f.y;
      tile[r][cc+2] = (bf16)f.z; tile[r][cc+3] = (bf16)f.w;
    }
  }
  __syncthreads();
  Frag16 w;
  #pragma unroll
  for (int i = 0; i < 16; ++i) w.v[i] = tile[c + i][r];
  v4u* q = (v4u*)(o + (size_t)(k0 + r) * NDM + n0 + c);
  q[0] = w.u[0]; q[1] = w.u[1];
}

// ---------------------------------------------------------------------------
// Setup 0c: RoPE cos/sin tables [s][d0..63].
// ---------------------------------------------------------------------------
__global__ __launch_bounds__(256)
void rope_tab(float* __restrict__ cosT, float* __restrict__ sinT) {
  int i = blockIdx.x * 256 + threadIdx.x;
  int s = i >> 6, d = i & 63;
  float inv = exp2f(-13.287712379549449f * ((float)d * (1.0f / 64.0f)));
  float th  = (float)s * inv;
  cosT[i] = cosf(th);
  sinT[i] = sinf(th);
}

// ---------------------------------------------------------------------------
// Kernel 1: fused QKV projection (bf16 WMMA) + bias + table-RoPE.
// A/B tiles DMA'd by the Tensor Data Mover into double-buffered LDS
// (fallback: register-staged copies). grid (16, 32, 3) x 256.
// ---------------------------------------------------------------------------
__global__ __launch_bounds__(256)
void qkv_rope_gemm(const unsigned short* __restrict__ hsb,
                   const unsigned short* __restrict__ Wt,
                   const float* __restrict__ bq, const float* __restrict__ bk,
                   const float* __restrict__ bv,
                   const float* __restrict__ cosT, const float* __restrict__ sinT,
                   unsigned short* __restrict__ Qo,
                   unsigned short* __restrict__ Kt,
                   unsigned short* __restrict__ Vo)
{
  const int which = blockIdx.z;
  const float* __restrict__ bias = (which == 0) ? bq : (which == 1) ? bk : bv;
  const unsigned short* __restrict__ Wp = Wt + (size_t)which * NDM * NDM;
  const int n0 = blockIdx.x * 128;
  const int m0 = blockIdx.y * 128;

  __shared__ __align__(16) bf16 lA[2][128][40];   // act [m][k], 64B rows + 16B pad
  __shared__ __align__(16) bf16 lB[2][32][136];   // Wt  [k][n], 256B rows + 16B pad

  const int tid  = threadIdx.x;
  const int lane = tid & 31;
  const int wave = tid >> 5;
  const int col   = lane & 15;
  const int rhalf = (lane >> 4) * 8;
  const int kb    = (lane >> 4) << 3;

  v8f acc[8];
  #pragma unroll
  for (int t = 0; t < 8; ++t)
    #pragma unroll
    for (int i = 0; i < 8; ++i) acc[t][i] = 0.f;

  const int NK = NDM / 32;                           // 64 k-steps

#if HAVE_TDM
  const unsigned short* aTile = hsb + (size_t)m0 * NDM;        // +i*32 along k
  const unsigned short* bTile = Wp + n0;                       // +i*32*NDM along k
  const unsigned ldsA[2] = { (unsigned)(size_t)&lA[0][0][0],
                             (unsigned)(size_t)&lA[1][0][0] };
  const unsigned ldsB[2] = { (unsigned)(size_t)&lB[0][0][0],
                             (unsigned)(size_t)&lB[1][0][0] };
  if (tid < 32) {                                    // wave 0 drives the TDM
    tdm_load_2d(ldsA[0], aTile, 32, 128, NDM, 3, 3); // 32-el rows (16 DW) + 4 DW pad
    tdm_load_2d(ldsB[0], bTile, 128, 32, NDM, 5, 3); // 128-el rows (64 DW) + 4 DW pad
  }
  for (int i = 0; i < NK; ++i) {
    if (tid < 32) __builtin_amdgcn_s_wait_tensorcnt(0);
    __syncthreads();                                 // tile i visible, buf[1-p] free
    if (i + 1 < NK && tid < 32) {
      const int nb = (i + 1) & 1;
      tdm_load_2d(ldsA[nb], aTile + (size_t)(i + 1) * 32,        32, 128, NDM, 3, 3);
      tdm_load_2d(ldsB[nb], bTile + (size_t)(i + 1) * 32 * NDM, 128,  32, NDM, 5, 3);
    }
    const int buf = i & 1;
    const int mr = wave * 16 + col;
    Frag16 fa;
    fa.u[0] = *(const v4u*)&lA[buf][mr][kb];
    fa.u[1] = *(const v4u*)&lA[buf][mr][16 + kb];
    #pragma unroll
    for (int t = 0; t < 8; ++t) {
      Frag16 fb;
      #pragma unroll
      for (int e = 0; e < 8; ++e) {
        fb.v[e]     = lB[buf][kb + e][t * 16 + col];
        fb.v[8 + e] = lB[buf][16 + kb + e][t * 16 + col];
      }
      acc[t] = __builtin_amdgcn_wmma_f32_16x16x32_bf16(
          false, fa.v, false, fb.v, (short)0, acc[t], false, false);
    }
  }
#else
  const int arow = tid >> 1, acol = (tid & 1) * 16;
  const int brow = tid >> 3, bcol = (tid & 7) * 16;
  const unsigned short* aptr = hsb + (size_t)(m0 + arow) * NDM + acol;
  const unsigned short* bptr = Wp + (size_t)brow * NDM + n0 + bcol;
  v4u ra0, ra1, rb0, rb1;
  { const v4u* p = (const v4u*)aptr; ra0 = p[0]; ra1 = p[1]; }
  { const v4u* p = (const v4u*)bptr; rb0 = p[0]; rb1 = p[1]; }
  *(v4u*)&lA[0][arow][acol] = ra0; *(v4u*)&lA[0][arow][acol + 8] = ra1;
  *(v4u*)&lB[0][brow][bcol] = rb0; *(v4u*)&lB[0][brow][bcol + 8] = rb1;
  for (int i = 0; i < NK; ++i) {
    if (i + 1 < NK) {
      const v4u* pa = (const v4u*)(aptr + (i + 1) * 32);
      const v4u* pb = (const v4u*)(bptr + (size_t)(i + 1) * 32 * NDM);
      ra0 = pa[0]; ra1 = pa[1]; rb0 = pb[0]; rb1 = pb[1];
    }
    __syncthreads();
    const int buf = i & 1;
    const int mr = wave * 16 + col;
    Frag16 fa;
    fa.u[0] = *(const v4u*)&lA[buf][mr][kb];
    fa.u[1] = *(const v4u*)&lA[buf][mr][16 + kb];
    #pragma unroll
    for (int t = 0; t < 8; ++t) {
      Frag16 fb;
      #pragma unroll
      for (int e = 0; e < 8; ++e) {
        fb.v[e]     = lB[buf][kb + e][t * 16 + col];
        fb.v[8 + e] = lB[buf][16 + kb + e][t * 16 + col];
      }
      acc[t] = __builtin_amdgcn_wmma_f32_16x16x32_bf16(
          false, fa.v, false, fb.v, (short)0, acc[t], false, false);
    }
    if (i + 1 < NK) {
      const int nb = (i + 1) & 1;
      *(v4u*)&lA[nb][arow][acol] = ra0; *(v4u*)&lA[nb][arow][acol + 8] = ra1;
      *(v4u*)&lB[nb][brow][bcol] = rb0; *(v4u*)&lB[nb][brow][bcol + 8] = rb1;
    }
  }
#endif

  // ----- epilogue: bias, RoPE via tables, scatter -----
  const int h = blockIdx.x;
  #pragma unroll
  for (int t = 0; t < 8; ++t) {
    float bv_ = bias[n0 + t * 16 + col];
    #pragma unroll
    for (int r = 0; r < 8; ++r) acc[t][r] += bv_;
  }

  if (which == 2) {                                  // V: (B,H,S,HD)
    #pragma unroll
    for (int t = 0; t < 8; ++t) {
      int d = t * 16 + col;
      #pragma unroll
      for (int r = 0; r < 8; ++r) {
        int m = m0 + wave * 16 + rhalf + r;
        int b = m >> 11, s = m & (NS - 1);
        Vo[((((size_t)b * NH + h) * NS) + s) * NHD + d] = f2bu(acc[t][r]);
      }
    }
  } else {
    #pragma unroll
    for (int t = 0; t < 4; ++t) {
      int d1 = t * 16 + col;                         // pair tile t <-> t+4
      #pragma unroll
      for (int r = 0; r < 8; ++r) {
        int m = m0 + wave * 16 + rhalf + r;
        int b = m >> 11, s = m & (NS - 1);
        float cs = cosT[s * 64 + d1], sn = sinT[s * 64 + d1];
        float y1 = acc[t][r], y2 = acc[t + 4][r];
        float o1 = y1 * cs - y2 * sn;
        float o2 = y2 * cs + y1 * sn;
        size_t bh = (size_t)b * NH + h;
        if (which == 0) {                            // Q: (B,H,S,HD)
          size_t rowi = (bh * NS + s) * NHD;
          Qo[rowi + d1]      = f2bu(o1);
          Qo[rowi + d1 + 64] = f2bu(o2);
        } else {                                     // K: transposed (B,H,HD,S)
          size_t base = bh * NHD;
          Kt[(base + d1)      * NS + s] = f2bu(o1);
          Kt[(base + d1 + 64) * NS + s] = f2bu(o2);
        }
      }
    }
  }
}

// ---------------------------------------------------------------------------
// Kernel 2: flash attention, register-staged K/V prefetch.
// grid (16, 16, 2) x 256.
// ---------------------------------------------------------------------------
__global__ __launch_bounds__(256)
void flash_attn(const unsigned short* __restrict__ Qo,
                const unsigned short* __restrict__ Kt,
                const unsigned short* __restrict__ Vo,
                const float* __restrict__ mask,
                float* __restrict__ out)
{
  const int q0 = blockIdx.x * 128;
  const int h  = blockIdx.y;
  const int b  = blockIdx.z;

  __shared__ __align__(16) bf16 lK[128][72];
  __shared__ __align__(16) bf16 lV[64][136];
  __shared__ __align__(16) bf16 lP[8][16][72];

  const int tid  = threadIdx.x;
  const int lane = tid & 31;
  const int wave = tid >> 5;
  const int col   = lane & 15;
  const int rhalf = (lane >> 4) * 8;
  const int kb    = (lane >> 4) << 3;

  const size_t bh = (size_t)b * NH + h;
  const unsigned short* Qp = Qo + bh * NS * NHD;
  const unsigned short* Kp = Kt + bh * NHD * NS;
  const unsigned short* Vp = Vo + bh * NS * NHD;

  const int kd = tid >> 1, kcb = (tid & 1) * 32;
  const int vj = tid >> 2, vcb = (tid & 3) * 32;
  const unsigned short* kptr = Kp + (size_t)kd * NS + kcb;
  const unsigned short* vptr = Vp + (size_t)vj * NHD + vcb;

  v16bf qf[4];
  {
    const unsigned short* qrow = Qp + (size_t)(q0 + wave * 16 + col) * NHD;
    #pragma unroll
    for (int c = 0; c < 4; ++c) {
      Frag16 f;
      f.u[0] = *(const v4u*)(qrow + c * 32 + kb);
      f.u[1] = *(const v4u*)(qrow + c * 32 + 16 + kb);
      qf[c] = f.v;
    }
  }

  v8f accO[8];
  #pragma unroll
  for (int t = 0; t < 8; ++t)
    #pragma unroll
    for (int i = 0; i < 8; ++i) accO[t][i] = 0.f;
  float mrun[8], lrun[8];
  #pragma unroll
  for (int r = 0; r < 8; ++r) { mrun[r] = -3.0e38f; lrun[r] = 0.f; }

  const float scale = 0.08838834764831845f;

  v4u kr[4], vr[4];
  { const v4u* g = (const v4u*)kptr;
    kr[0]=g[0]; kr[1]=g[1]; kr[2]=g[2]; kr[3]=g[3]; }
  { const v4u* g = (const v4u*)vptr;
    vr[0]=g[0]; vr[1]=g[1]; vr[2]=g[2]; vr[3]=g[3]; }

  for (int j0 = 0; j0 < NS; j0 += 64) {
    __syncthreads();
    { v4u* sp = (v4u*)&lK[kd][kcb];
      sp[0]=kr[0]; sp[1]=kr[1]; sp[2]=kr[2]; sp[3]=kr[3]; }
    { v4u* sp = (v4u*)&lV[vj][vcb];
      sp[0]=vr[0]; sp[1]=vr[1]; sp[2]=vr[2]; sp[3]=vr[3]; }
    if (j0 + 64 < NS) {
      const v4u* g = (const v4u*)(kptr + j0 + 64);
      kr[0]=g[0]; kr[1]=g[1]; kr[2]=g[2]; kr[3]=g[3];
      const v4u* g2 = (const v4u*)(vptr + (size_t)(j0 + 64) * NHD);
      vr[0]=g2[0]; vr[1]=g2[1]; vr[2]=g2[2]; vr[3]=g2[3];
    }
    __syncthreads();

    v8f sc[4];
    #pragma unroll
    for (int t = 0; t < 4; ++t)
      #pragma unroll
      for (int i = 0; i < 8; ++i) sc[t][i] = 0.f;
    #pragma unroll
    for (int t = 0; t < 4; ++t)
      #pragma unroll
      for (int c = 0; c < 4; ++c) {
        Frag16 fb;
        #pragma unroll
        for (int e = 0; e < 8; ++e) {
          fb.v[e]     = lK[c * 32 + kb + e][t * 16 + col];
          fb.v[8 + e] = lK[c * 32 + 16 + kb + e][t * 16 + col];
        }
        sc[t] = __builtin_amdgcn_wmma_f32_16x16x32_bf16(
            false, qf[c], false, fb.v, (short)0, sc[t], false, false);
      }

    float mnew[8];
    #pragma unroll
    for (int r = 0; r < 8; ++r) mnew[r] = mrun[r];
    #pragma unroll
    for (int t = 0; t < 4; ++t) {
      float mk = mask[(size_t)b * NS + j0 + t * 16 + col];
      #pragma unroll
      for (int r = 0; r < 8; ++r) {
        sc[t][r] = sc[t][r] * scale + mk;
        mnew[r]  = fmaxf(mnew[r], sc[t][r]);
      }
    }
    #pragma unroll
    for (int r = 0; r < 8; ++r) mnew[r] = wmax16(mnew[r]);

    float fac[8];
    #pragma unroll
    for (int r = 0; r < 8; ++r) { fac[r] = __expf(mrun[r] - mnew[r]); mrun[r] = mnew[r]; }
    #pragma unroll
    for (int t = 0; t < 8; ++t)
      #pragma unroll
      for (int r = 0; r < 8; ++r) accO[t][r] *= fac[r];

    float psum[8];
    #pragma unroll
    for (int r = 0; r < 8; ++r) psum[r] = 0.f;
    #pragma unroll
    for (int t = 0; t < 4; ++t)
      #pragma unroll
      for (int r = 0; r < 8; ++r) {
        float p = __expf(sc[t][r] - mnew[r]);
        sc[t][r] = p;
        psum[r] += p;
      }
    #pragma unroll
    for (int r = 0; r < 8; ++r) lrun[r] = lrun[r] * fac[r] + wsum16(psum[r]);

    #pragma unroll
    for (int t = 0; t < 4; ++t)
      #pragma unroll
      for (int r = 0; r < 8; ++r)
        lP[wave][rhalf + r][t * 16 + col] = (bf16)sc[t][r];
#if __has_builtin(__builtin_amdgcn_wave_barrier)
    __builtin_amdgcn_wave_barrier();
#endif

    #pragma unroll
    for (int c = 0; c < 2; ++c) {
      Frag16 fp;
      fp.u[0] = *(const v4u*)&lP[wave][col][c * 32 + kb];
      fp.u[1] = *(const v4u*)&lP[wave][col][c * 32 + 16 + kb];
      #pragma unroll
      for (int t = 0; t < 8; ++t) {
        Frag16 fb;
        #pragma unroll
        for (int e = 0; e < 8; ++e) {
          fb.v[e]     = lV[c * 32 + kb + e][t * 16 + col];
          fb.v[8 + e] = lV[c * 32 + 16 + kb + e][t * 16 + col];
        }
        accO[t] = __builtin_amdgcn_wmma_f32_16x16x32_bf16(
            false, fp.v, false, fb.v, (short)0, accO[t], false, false);
      }
    }
  }

  #pragma unroll
  for (int r = 0; r < 8; ++r) lrun[r] = 1.0f / lrun[r];
  #pragma unroll
  for (int t = 0; t < 8; ++t) {
    int d = t * 16 + col;
    #pragma unroll
    for (int r = 0; r < 8; ++r) {
      int s = q0 + wave * 16 + rhalf + r;
      out[((size_t)b * NS + s) * NDM + h * NHD + d] = accO[t][r] * lrun[r];
    }
  }
}

// ---------------------------------------------------------------------------
extern "C" void kernel_launch(void* const* d_in, const int* in_sizes, int n_in,
                              void* d_out, int out_size, void* d_ws, size_t ws_size,
                              hipStream_t stream) {
  (void)in_sizes; (void)n_in; (void)out_size; (void)ws_size;
  const float* hs   = (const float*)d_in[0];
  const float* mask = (const float*)d_in[1];
  const float* Wq   = (const float*)d_in[2];
  const float* bq   = (const float*)d_in[3];
  const float* Wk   = (const float*)d_in[4];
  const float* bk   = (const float*)d_in[5];
  const float* Wv   = (const float*)d_in[6];
  const float* bv   = (const float*)d_in[7];

  const size_t qkvE = (size_t)NB * NH * NS * NHD;
  const size_t wE   = (size_t)NDM * NDM;
  unsigned short* Qo  = (unsigned short*)d_ws;
  unsigned short* Kt  = Qo  + qkvE;
  unsigned short* Vo  = Kt  + qkvE;
  unsigned short* hsb = Vo  + qkvE;
  unsigned short* Wtb = hsb + (size_t)NM * NDM;
  float* cosT = (float*)(Wtb + 3 * wE);
  float* sinT = cosT + (size_t)NS * 64;

  cvt_hs  <<<dim3(4096, 1, 1), dim3(256,1,1), 0, stream>>>(hs, hsb);
  cvt_w   <<<dim3(32, 32, 3),  dim3(256,1,1), 0, stream>>>(Wq, Wk, Wv, Wtb);
  rope_tab<<<dim3(512, 1, 1),  dim3(256,1,1), 0, stream>>>(cosT, sinT);

  qkv_rope_gemm<<<dim3(NDM/128, NM/128, 3), dim3(256,1,1), 0, stream>>>(
      hsb, Wtb, bq, bk, bv, cosT, sinT, Qo, Kt, Vo);
  flash_attn<<<dim3(NS/128, NH, NB), dim3(256,1,1), 0, stream>>>(
      Qo, Kt, Vo, mask, (float*)d_out);
}
